// FixFeatureAttention_1529008358156
// MI455X (gfx1250) — matrix operand order repaired and checked by reading, not compiled, not run on hardware
//
#include <hip/hip_runtime.h>
#include <hip/hip_bf16.h>

typedef __attribute__((ext_vector_type(16))) _Float16 v16h;
typedef __attribute__((ext_vector_type(8)))  _Float16 v8h;
typedef __attribute__((ext_vector_type(8)))  float    v8f;

#define CH     512
#define INNER  256
#define HEADS  8
#define DHEAD  32
#define HW     4096
#define PROW   72   // padded LDS row stride (halves); 144B, 16B-aligned

// ---------------- WMMA helpers ----------------

__device__ __forceinline__ v8f wmma16(v16h a, v16h b, v8f c) {
  return __builtin_amdgcn_wmma_f32_16x16x32_f16(false, a, false, b, (short)0, c,
                                                false, false);
}

// A-matrix 16x32 f16 frag: lane m(<16) row m holds K {0..7,16..23};
// lane m+16 same row holds K {8..15,24..31}. Two 16B chunks per lane.
__device__ __forceinline__ v16h load_frag_a(const _Float16* m, int ld, int k0) {
  int lane = threadIdx.x & 31;
  const _Float16* p = m + (size_t)(lane & 15) * ld + k0 + ((lane < 16) ? 0 : 8);
  v8h lo = *(const v8h*)p;
  v8h hi = *(const v8h*)(p + 16);
  v16h r;
#pragma unroll
  for (int i = 0; i < 8; ++i) { r[i] = lo[i]; r[i + 8] = hi[i]; }
  return r;
}

// B-matrix 32x16 f16 frag: lane n(<16) = column n, K {0..7,16..23};
// lane n+16 same column, K {8..15,24..31}. Source stored column-per-row (row = N index).
__device__ __forceinline__ v16h load_frag_b(const _Float16* m, int ld, int k0) {
  int lane = threadIdx.x & 31;
  const _Float16* p = m + (size_t)(lane & 15) * ld + k0 + ((lane < 16) ? 0 : 8);
  v8h lo = *(const v8h*)p;
  v8h hi = *(const v8h*)(p + 16);
  v16h r;
#pragma unroll
  for (int i = 0; i < 8; ++i) { r[i] = lo[i]; r[i + 8] = hi[i]; }
  return r;
}

__device__ __forceinline__ float redmax16(float v) {
#pragma unroll
  for (int m = 8; m >= 1; m >>= 1) v = fmaxf(v, __shfl_xor(v, m, 32));
  return v;
}
__device__ __forceinline__ float redsum16(float v) {
#pragma unroll
  for (int m = 8; m >= 1; m >>= 1) v += __shfl_xor(v, m, 32);
  return v;
}

// ---------------- conversion kernels ----------------

__global__ void cvt_f16(const float* __restrict__ s, _Float16* __restrict__ d, int n) {
  int i = blockIdx.x * blockDim.x + threadIdx.x;
  if (i < n) d[i] = (_Float16)s[i];
}

// x [C][HW] f32 -> xt [HW][C] f16 (transpose so B-frags load rows)
__global__ void cvt_xT(const float* __restrict__ x, _Float16* __restrict__ xt) {
  int i = blockIdx.x * blockDim.x + threadIdx.x;  // HW*CH
  int p = i >> 9, c = i & (CH - 1);
  xt[i] = (_Float16)x[(size_t)c * HW + p];
}

// bilinear 16x16 -> 64x64, half-pixel centers (align_corners=False)
__global__ void pos_resize(const float* __restrict__ pb, float* __restrict__ pr) {
  int idx = blockIdx.x * blockDim.x + threadIdx.x;  // HEADS*HW
  int h = idx >> 12, p = idx & (HW - 1);
  int yy = p >> 6, xx = p & 63;
  float sy = (yy + 0.5f) * 0.25f - 0.5f;
  float sx = (xx + 0.5f) * 0.25f - 0.5f;
  float fy = sy - floorf(sy), fx = sx - floorf(sx);
  int y0 = (int)floorf(sy), x0 = (int)floorf(sx);
  int y1 = min(y0 + 1, 15), x1 = min(x0 + 1, 15);
  y0 = max(y0, 0); x0 = max(x0, 0);
  const float* s = pb + h * 256;
  float v00 = s[y0 * 16 + x0], v01 = s[y0 * 16 + x1];
  float v10 = s[y1 * 16 + x0], v11 = s[y1 * 16 + x1];
  float a = v00 + (v01 - v00) * fx;
  float b = v10 + (v11 - v10) * fx;
  pr[idx] = a + (b - a) * fy;
}

// ---------------- QKV projection GEMM (WMMA, 16x64 strip, SW-pipelined) ------
// out[o,p] = sum_c w[o,c]*x[c,p]; o in [0,768), p in [0,4096)
// q stored [h][p][d] pre-scaled, k stored [h][p][d], v stored transposed [h][d][p]
__global__ void __launch_bounds__(256) qkv_gemm(const _Float16* __restrict__ w,
                                                const _Float16* __restrict__ xt,
                                                _Float16* __restrict__ q,
                                                _Float16* __restrict__ k,
                                                _Float16* __restrict__ vt) {
  int wave = threadIdx.x >> 5, lane = threadIdx.x & 31;
  int tile = blockIdx.x * 8 + wave;            // 3072 strips: 48 x 64
  int o0 = (tile >> 6) << 4;                   // 48 strips of 16 rows
  int p0 = (tile & 63) << 6;                   // 64 strips of 64 pixels
  const _Float16* wb = w + (size_t)o0 * CH;
  const _Float16* x0 = xt + (size_t)p0 * CH;
  const _Float16* x1 = x0 + (size_t)16 * CH;
  const _Float16* x2 = x0 + (size_t)32 * CH;
  const _Float16* x3 = x0 + (size_t)48 * CH;
  v8f acc0 = {}, acc1 = {}, acc2 = {}, acc3 = {};
  // prologue loads
  v16h a  = load_frag_a(wb, CH, 0);
  v16h b0 = load_frag_b(x0, CH, 0);
  v16h b1 = load_frag_b(x1, CH, 0);
  v16h b2 = load_frag_b(x2, CH, 0);
  v16h b3 = load_frag_b(x3, CH, 0);
  for (int c = 0; c < CH; c += 32) {
    int cn = (c + 32) & (CH - 1);  // branchless wrap; last prefetch is dead
    v16h an  = load_frag_a(wb, CH, cn);
    v16h bn0 = load_frag_b(x0, CH, cn);
    v16h bn1 = load_frag_b(x1, CH, cn);
    v16h bn2 = load_frag_b(x2, CH, cn);
    v16h bn3 = load_frag_b(x3, CH, cn);
    acc0 = wmma16(a, b0, acc0);
    acc1 = wmma16(a, b1, acc1);
    acc2 = wmma16(a, b2, acc2);
    acc3 = wmma16(a, b3, acc3);
    a = an; b0 = bn0; b1 = bn1; b2 = bn2; b3 = bn3;
  }
  int col = lane & 15, rbase = (lane < 16) ? 0 : 8;
  const float scale = 0.17677669529663687f;    // 1/sqrt(32)
  v8f acc[4] = {acc0, acc1, acc2, acc3};
#pragma unroll
  for (int t = 0; t < 4; ++t) {
#pragma unroll
    for (int v = 0; v < 8; ++v) {
      int o = o0 + rbase + v, p = p0 + 16 * t + col;
      float val = acc[t][v];
      if (o < INNER) {
        int hh = o >> 5, d = o & 31;
        q[(size_t)hh * HW * DHEAD + (size_t)p * DHEAD + d] = (_Float16)(val * scale);
      } else if (o < 2 * INNER) {
        int oo = o - INNER, hh = oo >> 5, d = oo & 31;
        k[(size_t)hh * HW * DHEAD + (size_t)p * DHEAD + d] = (_Float16)val;
      } else {
        int oo = o - 2 * INNER, hh = oo >> 5, d = oo & 31;
        vt[(size_t)hh * DHEAD * HW + (size_t)d * HW + p] = (_Float16)val;
      }
    }
  }
}

// ---------------- flash attention (WMMA, online softmax, 64 keys/pass) --------
// grid (32, 8): blockIdx.x = query block of 128, blockIdx.y = head.
// Each wave: 16 queries. pos[i] cancels in softmax; only pos[j] is applied.
// K fragments are software-pipelined: next tile's loads issue under softmax.
__global__ void __launch_bounds__(256) attn_kernel(const _Float16* __restrict__ q,
                                                   const _Float16* __restrict__ k,
                                                   const _Float16* __restrict__ vt,
                                                   const float* __restrict__ posr,
                                                   _Float16* __restrict__ yt) {
  __shared__ _Float16 smP[8][16 * PROW];  // per-wave 16x64 P staging, padded rows
  int lane = threadIdx.x & 31, wave = threadIdx.x >> 5;
  int h = blockIdx.y;
  int q0 = blockIdx.x * 128 + wave * 16;
  const _Float16* qh = q + (size_t)h * HW * DHEAD;
  const _Float16* kh = k + (size_t)h * HW * DHEAD;
  const _Float16* vh = vt + (size_t)h * DHEAD * HW;
  const float* ph = posr + h * HW;

  v16h aq = load_frag_a(qh + (size_t)q0 * DHEAD, DHEAD, 0);
  v8f o0 = {}, o1 = {};
  float m[8], l[8];
#pragma unroll
  for (int v = 0; v < 8; ++v) { m[v] = -3.0e38f; l[v] = 0.f; }
  int col = lane & 15, rbase = (lane < 16) ? 0 : 8;
  _Float16* pbuf = &smP[wave][0];
  const v8f zc = {};

  // prologue K loads
  v16h bk0 = load_frag_b(kh,                        DHEAD, 0);
  v16h bk1 = load_frag_b(kh + (size_t)16 * DHEAD,   DHEAD, 0);
  v16h bk2 = load_frag_b(kh + (size_t)32 * DHEAD,   DHEAD, 0);
  v16h bk3 = load_frag_b(kh + (size_t)48 * DHEAD,   DHEAD, 0);

  for (int kt = 0; kt < HW; kt += 64) {
    v8f s0 = wmma16(aq, bk0, zc);
    v8f s1 = wmma16(aq, bk1, zc);
    v8f s2 = wmma16(aq, bk2, zc);
    v8f s3 = wmma16(aq, bk3, zc);
    // prefetch next key tile; latency hides under the softmax VALU phase
    int ktn = (kt + 64) & (HW - 1);  // branchless wrap; last prefetch is dead
    v16h bkn0 = load_frag_b(kh + (size_t)(ktn     ) * DHEAD, DHEAD, 0);
    v16h bkn1 = load_frag_b(kh + (size_t)(ktn + 16) * DHEAD, DHEAD, 0);
    v16h bkn2 = load_frag_b(kh + (size_t)(ktn + 32) * DHEAD, DHEAD, 0);
    v16h bkn3 = load_frag_b(kh + (size_t)(ktn + 48) * DHEAD, DHEAD, 0);
    float pj0 = ph[kt + col];
    float pj1 = ph[kt + 16 + col];
    float pj2 = ph[kt + 32 + col];
    float pj3 = ph[kt + 48 + col];
#pragma unroll
    for (int v = 0; v < 8; ++v) {
      float a0 = s0[v] + pj0;
      float a1 = s1[v] + pj1;
      float a2 = s2[v] + pj2;
      float a3 = s3[v] + pj3;
      float rm = redmax16(fmaxf(fmaxf(a0, a1), fmaxf(a2, a3)));
      float mn = fmaxf(m[v], rm);
      float alpha = __expf(m[v] - mn);
      m[v] = mn;
      float p0 = __expf(a0 - mn);
      float p1 = __expf(a1 - mn);
      float p2 = __expf(a2 - mn);
      float p3 = __expf(a3 - mn);
      l[v] = l[v] * alpha + redsum16((p0 + p1) + (p2 + p3));
      o0[v] *= alpha;
      o1[v] *= alpha;
      _Float16* r = pbuf + (rbase + v) * PROW + col;
      r[0]  = (_Float16)p0;
      r[16] = (_Float16)p1;
      r[32] = (_Float16)p2;
      r[48] = (_Float16)p3;
    }
    // issue global V loads early so they overlap the LDS turnaround
    v16h bv00 = load_frag_b(vh, HW, kt);                        // d 0..15,  j kt..+31
    v16h bv01 = load_frag_b(vh, HW, kt + 32);                   // d 0..15,  j +32..+63
    v16h bv10 = load_frag_b(vh + (size_t)16 * HW, HW, kt);      // d 16..31
    v16h bv11 = load_frag_b(vh + (size_t)16 * HW, HW, kt + 32);
    // same-wave LDS is in-order; fence compiler + wait for stores to land
    asm volatile("s_wait_dscnt 0" ::: "memory");
    v16h ap0 = load_frag_a(pbuf, PROW, 0);    // P cols 0..31
    v16h ap1 = load_frag_a(pbuf, PROW, 32);   // P cols 32..63
    o0 = wmma16(ap0, bv00, o0);
    o0 = wmma16(ap1, bv01, o0);
    o1 = wmma16(ap0, bv10, o1);
    o1 = wmma16(ap1, bv11, o1);
    bk0 = bkn0; bk1 = bkn1; bk2 = bkn2; bk3 = bkn3;
  }
#pragma unroll
  for (int v = 0; v < 8; ++v) {
    float inv = 1.0f / l[v];
    int qq = q0 + rbase + v;
    yt[(size_t)qq * INNER + h * DHEAD + col]      = (_Float16)(o0[v] * inv);
    yt[(size_t)qq * INNER + h * DHEAD + col + 16] = (_Float16)(o1[v] * inv);
  }
}

// ---------------- projection + bias + residual (WMMA, 16x64, SW-pipelined) ---
__global__ void __launch_bounds__(256) proj_res(const _Float16* __restrict__ wp,
                                                const _Float16* __restrict__ yt,
                                                const float* __restrict__ x,
                                                const float* __restrict__ bp,
                                                float* __restrict__ z) {
  int wave = threadIdx.x >> 5, lane = threadIdx.x & 31;
  int tile = blockIdx.x * 8 + wave;  // 2048 strips: 32 x 64
  int c0 = (tile >> 6) << 4;
  int p0 = (tile & 63) << 6;
  const _Float16* wb = wp + (size_t)c0 * INNER;
  const _Float16* y0 = yt + (size_t)p0 * INNER;
  const _Float16* y1 = y0 + (size_t)16 * INNER;
  const _Float16* y2 = y0 + (size_t)32 * INNER;
  const _Float16* y3 = y0 + (size_t)48 * INNER;
  v8f acc0 = {}, acc1 = {}, acc2 = {}, acc3 = {};
  v16h a  = load_frag_a(wb, INNER, 0);
  v16h b0 = load_frag_b(y0, INNER, 0);
  v16h b1 = load_frag_b(y1, INNER, 0);
  v16h b2 = load_frag_b(y2, INNER, 0);
  v16h b3 = load_frag_b(y3, INNER, 0);
  for (int o = 0; o < INNER; o += 32) {
    int on = (o + 32) & (INNER - 1);  // branchless wrap
    v16h an  = load_frag_a(wb, INNER, on);
    v16h bn0 = load_frag_b(y0, INNER, on);
    v16h bn1 = load_frag_b(y1, INNER, on);
    v16h bn2 = load_frag_b(y2, INNER, on);
    v16h bn3 = load_frag_b(y3, INNER, on);
    acc0 = wmma16(a, b0, acc0);
    acc1 = wmma16(a, b1, acc1);
    acc2 = wmma16(a, b2, acc2);
    acc3 = wmma16(a, b3, acc3);
    a = an; b0 = bn0; b1 = bn1; b2 = bn2; b3 = bn3;
  }
  int col = lane & 15, rbase = (lane < 16) ? 0 : 8;
  v8f acc[4] = {acc0, acc1, acc2, acc3};
#pragma unroll
  for (int t = 0; t < 4; ++t) {
#pragma unroll
    for (int v = 0; v < 8; ++v) {
      int c = c0 + rbase + v, p = p0 + 16 * t + col;
      size_t i = (size_t)c * HW + p;
      z[i] = acc[t][v] + bp[c] + x[i];
    }
  }
}

// ---------------- channel LayerNorm ----------------
__global__ void ln_kernel(const float* __restrict__ z, const float* __restrict__ gamma,
                          const float* __restrict__ beta, float* __restrict__ out) {
  int p = blockIdx.x * blockDim.x + threadIdx.x;  // one pixel per thread
  float s = 0.f, s2 = 0.f;
  for (int c = 0; c < CH; ++c) {
    float t = z[(size_t)c * HW + p];
    s += t; s2 += t * t;
  }
  float mu = s * (1.0f / CH);
  float var = s2 * (1.0f / CH) - mu * mu;
  float inv = rsqrtf(var + 1e-5f);
  for (int c = 0; c < CH; ++c) {
    float t = z[(size_t)c * HW + p];
    out[(size_t)c * HW + p] = (t - mu) * inv * gamma[c] + beta[c];
  }
}

// ---------------- launcher ----------------

extern "C" void kernel_launch(void* const* d_in, const int* in_sizes, int n_in,
                              void* d_out, int out_size, void* d_ws, size_t ws_size,
                              hipStream_t stream) {
  const float* x        = (const float*)d_in[0];
  const float* w_qkv    = (const float*)d_in[1];
  const float* w_proj   = (const float*)d_in[2];
  const float* b_proj   = (const float*)d_in[3];
  const float* pos_bias = (const float*)d_in[4];
  const float* gamma    = (const float*)d_in[5];
  const float* beta     = (const float*)d_in[6];
  float* out = (float*)d_out;

  char* ws = (char*)d_ws;
  _Float16* xt  = (_Float16*)ws; ws += (size_t)HW * CH * 2;          // 4 MB
  _Float16* wqh = (_Float16*)ws; ws += (size_t)(3 * INNER) * CH * 2; // 0.75 MB
  _Float16* wph = (_Float16*)ws; ws += (size_t)CH * INNER * 2;       // 0.25 MB
  _Float16* qf  = (_Float16*)ws; ws += (size_t)INNER * HW * 2;       // 2 MB
  _Float16* kf  = (_Float16*)ws; ws += (size_t)INNER * HW * 2;       // 2 MB
  _Float16* vtf = (_Float16*)ws; ws += (size_t)INNER * HW * 2;       // 2 MB
  float*    posr = (float*)ws;   ws += (size_t)HEADS * HW * 4;       // 128 KB
  _Float16* ytf = (_Float16*)ws; ws += (size_t)HW * INNER * 2;       // 2 MB
  float*    z   = (float*)ws;    ws += (size_t)CH * HW * 4;          // 8 MB

  cvt_f16<<<(3 * INNER * CH) / 256, 256, 0, stream>>>(w_qkv, wqh, 3 * INNER * CH);
  cvt_f16<<<(CH * INNER) / 256, 256, 0, stream>>>(w_proj, wph, CH * INNER);
  cvt_xT<<<(HW * CH) / 256, 256, 0, stream>>>(x, xt);
  pos_resize<<<(HEADS * HW) / 256, 256, 0, stream>>>(pos_bias, posr);

  qkv_gemm<<<384, 256, 0, stream>>>(wqh, xt, qf, kf, vtf);
  attn_kernel<<<dim3(32, 8), 256, 0, stream>>>(qf, kf, vtf, posr, ytf);
  proj_res<<<256, 256, 0, stream>>>(wph, ytf, x, b_proj, z);
  ln_kernel<<<HW / 256, 256, 0, stream>>>(z, gamma, beta, out);
}
